// GeneralHetero_2680059592836
// MI455X (gfx1250) — matrix operand order repaired and checked by reading, not compiled, not run on hardware
//
#include <hip/hip_runtime.h>
#include <hip/hip_bf16.h>

#define N_USER 200000
#define N_ITEM 100000
#define N_EDGE 3200000
#define D      64
#define NLAYER 2
#define LN_EPS 1e-5f

typedef __attribute__((ext_vector_type(16))) __bf16 v16bf;
typedef __attribute__((ext_vector_type(8)))  float  v8f;

__device__ __forceinline__ unsigned pack2_bf16(float x, float y) {
    union { __bf16 b[2]; unsigned u; } c;
    c.b[0] = (__bf16)x;
    c.b[1] = (__bf16)y;
    return c.u;
}

union Frag {
    uint4  q[2];
    v16bf  v;
};

// ---------------------------------------------------------------------------
// zero fill
// ---------------------------------------------------------------------------
__global__ void zero_kernel(float* __restrict__ p, int n) {
    int i = blockIdx.x * blockDim.x + threadIdx.x;
    if (i < n) p[i] = 0.0f;
}

// ---------------------------------------------------------------------------
// edge scatter: agg[dst] += h[src], cnt[dst] += 1   (16 threads per edge)
// ---------------------------------------------------------------------------
__global__ void scatter_kernel(const float* __restrict__ hsrc,
                               const int*   __restrict__ src,
                               const int*   __restrict__ dst,
                               float* __restrict__ agg,
                               float* __restrict__ cnt) {
    long long t = (long long)blockIdx.x * blockDim.x + threadIdx.x;
    int e = (int)(t >> 4);
    int q = (int)(t & 15);
    if (e >= N_EDGE) return;
    int s = src[e];
    int d = dst[e];
    const float4 v = *(const float4*)(hsrc + (size_t)s * D + q * 4);
    float* base = agg + (size_t)d * D + q * 4;
    atomicAdd(base + 0, v.x);
    atomicAdd(base + 1, v.y);
    atomicAdd(base + 2, v.z);
    atomicAdd(base + 3, v.w);
    if (q == 0) atomicAdd(cnt + d, 1.0f);
}

// ---------------------------------------------------------------------------
// fused node update: out = relu(LN( (agg/cnt)@Wrel + h@Wroot + b + h ))
// 256 threads = 8 waves; block covers 128 nodes; wave computes a 16x64 tile.
// Tiles are staged in LDS *pre-swizzled into WMMA fragment layout* (packed
// bf16 pairs), so each fragment is two ds_load_b128 per lane.
// ---------------------------------------------------------------------------
__global__ __launch_bounds__(256) void update_kernel(
        int n_nodes,
        const float* __restrict__ agg,  const float* __restrict__ cnt,
        const float* __restrict__ h_in,
        const float* __restrict__ Wrel, const float* __restrict__ Wroot,
        const float* __restrict__ bias,
        const float* __restrict__ ln_g, const float* __restrict__ ln_b,
        float* __restrict__ h_out) {
    // [wave][ks][half][lane] : 16B per lane slot -> conflict-free b128 reads
    __shared__ uint4 aFragA[8][2][2][32];    // agg/cnt tile fragments (16 KB)
    __shared__ uint4 aFragH[8][2][2][32];    // root-feature tile fragments
    __shared__ uint4 bFragRel[2][4][2][32];  // [ks][nt][half][lane] (8 KB)
    __shared__ uint4 bFragRoot[2][4][2][32];
    __shared__ float sInv[128];

    const int tid  = threadIdx.x;
    const int lane = tid & 31;
    const int wave = tid >> 5;
    const int nb   = blockIdx.x * 128;

    // reciprocal counts (mean aggregation, isolated nodes -> /1)
    if (tid < 128) {
        int row = nb + tid;
        float c = (row < n_nodes) ? cnt[row] : 1.0f;
        sInv[tid] = 1.0f / fmaxf(c, 1.0f);
    }

    // ---- stage weights directly into B-fragment layout ----
    // element i=2v+p of lane (hi*16+colL) holds W[ks*32 + (v>>2)*16 + hi*8 + 2(v&3) + p][nt*16+colL]
    for (int t = tid; t < 2048; t += 256) {
        int colL = t & 15;
        int nt   = (t >> 4) & 3;
        int b    = t >> 6;              // 0..31
        int v    = b & 7;
        int hb   = (b >> 3) & 1;
        int ks   = (b >> 4) & 1;
        int col  = nt * 16 + colL;
        int kb   = ks * 32 + (v >> 2) * 16 + hb * 8 + 2 * (v & 3);
        unsigned prel = pack2_bf16(Wrel[(size_t)kb * D + col],
                                   Wrel[(size_t)(kb + 1) * D + col]);
        unsigned proo = pack2_bf16(Wroot[(size_t)kb * D + col],
                                   Wroot[(size_t)(kb + 1) * D + col]);
        int ln = hb * 16 + colL;
        ((unsigned*)&bFragRel[ks][nt][v >> 2][ln])[v & 3]  = prel;
        ((unsigned*)&bFragRoot[ks][nt][v >> 2][ln])[v & 3] = proo;
    }
    __syncthreads();   // sInv ready before A staging uses it

    // ---- stage node tiles directly into A-fragment layout ----
    // (r, c=2j..2j+1) -> lane = hi*16 + (r&15), dword v; coalesced float2 reads
    for (int t = tid; t < 128 * 32; t += 256) {
        int r = t >> 5;                 // 0..127
        int j = t & 31;                 // column pair, c = 2j
        int row = nb + r;
        float a0 = 0.f, a1 = 0.f, h0 = 0.f, h1 = 0.f;
        if (row < n_nodes) {
            const float2 av = *(const float2*)(agg  + (size_t)row * D + 2 * j);
            const float2 hv = *(const float2*)(h_in + (size_t)row * D + 2 * j);
            float inv = sInv[r];
            a0 = av.x * inv; a1 = av.y * inv;
            h0 = hv.x;       h1 = hv.y;
        }
        int ks = j >> 4;
        int jp = j & 15;                // kk = 2*jp
        int hb = (jp >> 2) & 1;         // kk bit3
        int vg = (jp >> 3) & 1;         // kk bit4
        int v  = vg * 4 + (jp & 3);
        int ln = hb * 16 + (r & 15);
        ((unsigned*)&aFragA[r >> 4][ks][v >> 2][ln])[v & 3] = pack2_bf16(a0, a1);
        ((unsigned*)&aFragH[r >> 4][ks][v >> 2][ln])[v & 3] = pack2_bf16(h0, h1);
    }
    __syncthreads();

    const int hi = lane >> 4;    // lane half
    const int ml = lane & 15;    // row (A) / col (B,C) within tile

    v8f acc[4];
    #pragma unroll
    for (int nt = 0; nt < 4; ++nt)
        #pragma unroll
        for (int i = 0; i < 8; ++i) acc[nt][i] = 0.0f;

    #pragma unroll
    for (int ks = 0; ks < 2; ++ks) {
        Frag aA, aH;
        aA.q[0] = aFragA[wave][ks][0][lane];
        aA.q[1] = aFragA[wave][ks][1][lane];
        aH.q[0] = aFragH[wave][ks][0][lane];
        aH.q[1] = aFragH[wave][ks][1][lane];
        #pragma unroll
        for (int nt = 0; nt < 4; ++nt) {
            Frag bR, bO;
            bR.q[0] = bFragRel[ks][nt][0][lane];
            bR.q[1] = bFragRel[ks][nt][1][lane];
            bO.q[0] = bFragRoot[ks][nt][0][lane];
            bO.q[1] = bFragRoot[ks][nt][1][lane];
            acc[nt] = __builtin_amdgcn_wmma_f32_16x16x32_bf16(
                false, aA.v, false, bR.v, (short)0, acc[nt], false, false);
            acc[nt] = __builtin_amdgcn_wmma_f32_16x16x32_bf16(
                false, aH.v, false, bO.v, (short)0, acc[nt], false, false);
        }
    }

    // residual + bias in f32 (C layout: VGPR r -> row = r + 8*hi, col = nt*16+ml)
    #pragma unroll
    for (int nt = 0; nt < 4; ++nt) {
        int col = nt * 16 + ml;
        float bv = bias[col];
        #pragma unroll
        for (int r = 0; r < 8; ++r) {
            int row = nb + wave * 16 + (hi ? r + 8 : r);
            float add = bv;
            if (row < n_nodes) add += h_in[(size_t)row * D + col];
            acc[nt][r] += add;
        }
    }

    // LayerNorm + ReLU per row; row stats reduced inside each 16-lane half
    #pragma unroll
    for (int r = 0; r < 8; ++r) {
        float s = acc[0][r] + acc[1][r] + acc[2][r] + acc[3][r];
        float q = acc[0][r] * acc[0][r] + acc[1][r] * acc[1][r] +
                  acc[2][r] * acc[2][r] + acc[3][r] * acc[3][r];
        #pragma unroll
        for (int m = 1; m < 16; m <<= 1) {
            s += __shfl_xor(s, m, 32);
            q += __shfl_xor(q, m, 32);
        }
        float mean = s * (1.0f / 64.0f);
        float var  = q * (1.0f / 64.0f) - mean * mean;
        float rstd = rsqrtf(var + LN_EPS);
        int row = nb + wave * 16 + (hi ? r + 8 : r);
        if (row < n_nodes) {
            #pragma unroll
            for (int nt = 0; nt < 4; ++nt) {
                int col = nt * 16 + ml;
                float y = (acc[nt][r] - mean) * rstd * ln_g[col] + ln_b[col];
                h_out[(size_t)row * D + col] = fmaxf(y, 0.0f);
            }
        }
    }
}

// ---------------------------------------------------------------------------
// driver
// ---------------------------------------------------------------------------
extern "C" void kernel_launch(void* const* d_in, const int* in_sizes, int n_in,
                              void* d_out, int out_size, void* d_ws, size_t ws_size,
                              hipStream_t stream) {
    const float* x_user     = (const float*)d_in[0];
    const float* x_item     = (const float*)d_in[1];
    const int*   src_u2i    = (const int*)  d_in[2];
    const int*   dst_u2i    = (const int*)  d_in[3];
    const int*   src_i2u    = (const int*)  d_in[4];
    const int*   dst_i2u    = (const int*)  d_in[5];
    const float* W_rel_u2i  = (const float*)d_in[6];
    const float* W_root_u2i = (const float*)d_in[7];
    const float* b_u2i      = (const float*)d_in[8];
    const float* W_rel_i2u  = (const float*)d_in[9];
    const float* W_root_i2u = (const float*)d_in[10];
    const float* b_i2u      = (const float*)d_in[11];
    const float* ln_g_user  = (const float*)d_in[12];
    const float* ln_b_user  = (const float*)d_in[13];
    const float* ln_g_item  = (const float*)d_in[14];
    const float* ln_b_item  = (const float*)d_in[15];

    float* ws    = (float*)d_ws;
    float* agg_u = ws;                                  // N_USER*D
    float* agg_i = agg_u + (size_t)N_USER * D;          // N_ITEM*D
    float* cnt_u = agg_i + (size_t)N_ITEM * D;          // N_USER
    float* cnt_i = cnt_u + (size_t)N_USER;              // N_ITEM
    float* hb_u  = cnt_i + (size_t)N_ITEM;              // N_USER*D
    float* hb_i  = hb_u  + (size_t)N_USER * D;          // N_ITEM*D

    float* out_u = (float*)d_out;
    float* out_i = out_u + (size_t)N_USER * D;

    const float* hu = x_user;
    const float* hi = x_item;

    const int ZB = 256;
    dim3 blk(256);
    dim3 g_zero_u(((size_t)N_USER * D + ZB - 1) / ZB);
    dim3 g_zero_i(((size_t)N_ITEM * D + ZB - 1) / ZB);
    dim3 g_zero_cu((N_USER + ZB - 1) / ZB);
    dim3 g_zero_ci((N_ITEM + ZB - 1) / ZB);
    dim3 g_scat((((long long)N_EDGE * 16) + ZB - 1) / ZB);
    dim3 g_up_i((N_ITEM + 127) / 128);
    dim3 g_up_u((N_USER + 127) / 128);

    for (int l = 0; l < NLAYER; ++l) {
        const bool last = (l == NLAYER - 1);
        float* nu = last ? out_u : hb_u;
        float* ni = last ? out_i : hb_i;

        zero_kernel<<<g_zero_i,  blk, 0, stream>>>(agg_i, N_ITEM * D);
        zero_kernel<<<g_zero_ci, blk, 0, stream>>>(cnt_i, N_ITEM);
        zero_kernel<<<g_zero_u,  blk, 0, stream>>>(agg_u, N_USER * D);
        zero_kernel<<<g_zero_cu, blk, 0, stream>>>(cnt_u, N_USER);

        scatter_kernel<<<g_scat, blk, 0, stream>>>(hu, src_u2i, dst_u2i, agg_i, cnt_i);
        scatter_kernel<<<g_scat, blk, 0, stream>>>(hi, src_i2u, dst_i2u, agg_u, cnt_u);

        update_kernel<<<g_up_i, blk, 0, stream>>>(
            N_ITEM, agg_i, cnt_i, hi,
            W_rel_u2i + (size_t)l * D * D, W_root_u2i + (size_t)l * D * D,
            b_u2i + (size_t)l * D, ln_g_item + (size_t)l * D, ln_b_item + (size_t)l * D,
            ni);
        update_kernel<<<g_up_u, blk, 0, stream>>>(
            N_USER, agg_u, cnt_u, hu,
            W_rel_i2u + (size_t)l * D * D, W_root_i2u + (size_t)l * D * D,
            b_i2u + (size_t)l * D, ln_g_user + (size_t)l * D, ln_b_user + (size_t)l * D,
            nu);

        hu = nu;
        hi = ni;
    }
}
